// LinearMem_61770219651188
// MI455X (gfx1250) — 8-bit matrix operand order repaired and checked by reading, compiled, not run on hardware
//
#include <hip/hip_runtime.h>
#include <stdint.h>

typedef int v8i __attribute__((ext_vector_type(8)));

#define M_TOK 2048
#define NF    2048
#define KF    2048
#define BM    128
#define BN    128
#define BK    128

__global__ __launch_bounds__(256) void absmax_kernel(const float* __restrict__ src,
                                                     unsigned n,
                                                     unsigned* __restrict__ out_bits) {
    __shared__ float red[256];
    unsigned i = blockIdx.x * 256u + threadIdx.x;
    unsigned stride = gridDim.x * 256u;
    float m = 0.0f;
    for (; i < n; i += stride) m = fmaxf(m, fabsf(src[i]));
    red[threadIdx.x] = m;
    __syncthreads();
    for (int s = 128; s > 0; s >>= 1) {
        if ((int)threadIdx.x < s)
            red[threadIdx.x] = fmaxf(red[threadIdx.x], red[threadIdx.x + s]);
        __syncthreads();
    }
    if (threadIdx.x == 0) atomicMax(out_bits, __float_as_uint(red[0]));
}

__device__ __forceinline__ int clamp_q(float v, float inv) {
    int q = __float2int_rn(v * inv);
    q = q > 127 ? 127 : q;
    q = q < -127 ? -127 : q;
    return q;
}

__global__ __launch_bounds__(256) void quant_kernel(const float* __restrict__ src,
                                                    signed char* __restrict__ dst,
                                                    const unsigned* __restrict__ max_bits,
                                                    int n16) {
    int i = blockIdx.x * 256 + threadIdx.x;
    if (i >= n16) return;
    float inv = 127.0f / __uint_as_float(*max_bits);
    const float4* s = (const float4*)src + (size_t)i * 4;
    unsigned p[4];
#pragma unroll
    for (int j = 0; j < 4; ++j) {
        float4 v = s[j];
        unsigned b0 = (unsigned)clamp_q(v.x, inv) & 255u;
        unsigned b1 = (unsigned)clamp_q(v.y, inv) & 255u;
        unsigned b2 = (unsigned)clamp_q(v.z, inv) & 255u;
        unsigned b3 = (unsigned)clamp_q(v.w, inv) & 255u;
        p[j] = b0 | (b1 << 8) | (b2 << 16) | (b3 << 24);
    }
    *(uint4*)(dst + (size_t)i * 16) = make_uint4(p[0], p[1], p[2], p[3]);
}

__device__ __forceinline__ unsigned lds_off(const void* p) {
    return (unsigned)(unsigned long long)(uintptr_t)p;
}
__device__ __forceinline__ void async_load_b128_s(unsigned lds, unsigned voff,
                                                  const void* sbase) {
    asm volatile("global_load_async_to_lds_b128 %0, %1, %2"
                 :: "v"(lds), "v"(voff), "s"((unsigned long long)(uintptr_t)sbase));
}
__device__ __forceinline__ void wait_async0() {
    asm volatile("s_wait_asynccnt 0x0" ::: "memory");
}

__global__ __launch_bounds__(256) void gemm_iu8_kernel(const signed char* __restrict__ qa,
                                                       const signed char* __restrict__ qb,
                                                       const float* __restrict__ bias,
                                                       const unsigned* __restrict__ scale_bits,
                                                       float* __restrict__ out) {
    __shared__ signed char As[2][BM * BK];
    __shared__ signed char Bs[2][BN * BK];

    const int tid  = threadIdx.x;
    const int lane = tid & 31;
    const int wave = tid >> 5;
    const int wm   = wave & 1;
    const int wn   = wave >> 1;
    const int half = lane >> 4;
    const int l15  = lane & 15;

    const int m_base = blockIdx.y * BM;
    const int n_base = blockIdx.x * BN;

    const int lrow = tid >> 3;
    const int lcol = (tid & 7) * 16;
    const unsigned voffA = (unsigned)((m_base + lrow) * KF + lcol);
    const unsigned voffB = (unsigned)((n_base + lrow) * KF + lcol);
    unsigned aL[2][4], bL[2][4];
#pragma unroll
    for (int b = 0; b < 2; ++b)
#pragma unroll
        for (int p = 0; p < 4; ++p) {
            aL[b][p] = lds_off(&As[b][(lrow + p * 32) * BK + lcol]);
            bL[b][p] = lds_off(&Bs[b][(lrow + p * 32) * BK + lcol]);
        }

    auto issue_tile = [&](int b, int ktile) {
        const size_t kb = (size_t)ktile * BK;
#pragma unroll
        for (int p = 0; p < 4; ++p) {
            async_load_b128_s(aL[b][p], voffA, qa + kb + (size_t)p * 32 * KF);
            async_load_b128_s(bL[b][p], voffB, qb + kb + (size_t)p * 32 * KF);
        }
    };

    v8i acc[4][2];
#pragma unroll
    for (int i = 0; i < 4; ++i)
#pragma unroll
        for (int j = 0; j < 2; ++j)
            acc[i][j] = (v8i){0, 0, 0, 0, 0, 0, 0, 0};

    auto compute_tile = [&](const signed char* __restrict__ Asb,
                            const signed char* __restrict__ Bsb) {
#pragma unroll
        for (int s = 0; s < 2; ++s) {
            v8i bfrag[2];
#pragma unroll
            for (int nt = 0; nt < 2; ++nt) {
                const signed char* bp = Bsb + (wn * 32 + nt * 16 + l15) * BK + s * 64;
                union { uint4 q[2]; v8i v; } f;
                f.q[0] = *(const uint4*)(bp + 16 * half);
                f.q[1] = *(const uint4*)(bp + 32 + 16 * half);
                bfrag[nt] = f.v;
            }
            v8i afrag[4];
#pragma unroll
            for (int mt = 0; mt < 4; ++mt) {
                const signed char* ap =
                    Asb + (wm * 64 + mt * 16 + l15) * BK + s * 64 + 16 * half;
                union { uint4 q[2]; v8i v; } f;
                f.q[0] = *(const uint4*)(ap + 0);
                f.q[1] = *(const uint4*)(ap + 32);
                afrag[mt] = f.v;
            }
#pragma unroll
            for (int mt = 0; mt < 4; ++mt)
#pragma unroll
                for (int nt = 0; nt < 2; ++nt)
                    acc[mt][nt] = __builtin_amdgcn_wmma_i32_16x16x64_iu8(
                         true, afrag[mt],  true, bfrag[nt],
                        acc[mt][nt],  false,  false);
        }
    };

    issue_tile(0, 0);

    const int NKT = KF / BK;
#pragma unroll
    for (int kt = 0; kt < NKT; kt += 2) {
        wait_async0();
        __syncthreads();
        if (kt + 1 < NKT) issue_tile(1, kt + 1);
        compute_tile(As[0], Bs[0]);

        wait_async0();
        __syncthreads();
        if (kt + 2 < NKT) issue_tile(0, kt + 2);
        compute_tile(As[1], Bs[1]);
    }

    const float sx = __uint_as_float(scale_bits[0]) * (1.0f / 127.0f);
    const float sw = __uint_as_float(scale_bits[1]) * (1.0f / 127.0f);
    const float scale = sx * sw;

#pragma unroll
    for (int mt = 0; mt < 4; ++mt) {
        const int m0 = m_base + wm * 64 + mt * 16 + 8 * half;
#pragma unroll
        for (int nt = 0; nt < 2; ++nt) {
            const int n = n_base + wn * 32 + nt * 16 + l15;
            const float b = bias[n];
#pragma unroll
            for (int r = 0; r < 8; ++r) {
                out[(size_t)(m0 + r) * NF + n] = (float)acc[mt][nt][r] * scale + b;
            }
        }
    }
}

extern "C" void kernel_launch(void* const* d_in, const int* in_sizes, int n_in,
                              void* d_out, int out_size, void* d_ws, size_t ws_size,
                              hipStream_t stream) {
    const float* x    = (const float*)d_in[0];
    const float* w    = (const float*)d_in[1];
    const float* bias = (const float*)d_in[2];
    float* out        = (float*)d_out;

    unsigned char* ws = (unsigned char*)d_ws;
    unsigned* bits    = (unsigned*)ws;
    signed char* qx   = (signed char*)(ws + 256);
    signed char* qw   = qx + (size_t)M_TOK * KF;

    (void)in_sizes; (void)n_in; (void)out_size; (void)ws_size;

    hipMemsetAsync(d_ws, 0, 8, stream);

    const unsigned n = (unsigned)M_TOK * KF;
    absmax_kernel<<<512, 256, 0, stream>>>(x, n, bits);
    absmax_kernel<<<512, 256, 0, stream>>>(w, n, bits + 1);

    const int n16 = (int)(n / 16);
    quant_kernel<<<n16 / 256, 256, 0, stream>>>(x, qx, bits, n16);
    quant_kernel<<<n16 / 256, 256, 0, stream>>>(w, qw, bits + 1, n16);

    dim3 grid(NF / BN, M_TOK / BM);
    gemm_iu8_kernel<<<grid, 256, 0, stream>>>(qx, qw, bias, bits, out);
}
